// AttributeBranch_87969520157431
// MI455X (gfx1250) — matrix operand order repaired and checked by reading, not compiled, not run on hardware
//
#include <hip/hip_runtime.h>

#define DD   256   // sketch dim
#define CC   256   // channels
#define BB   64    // batch
#define HW   784   // 28*28  (= 49 * 16, so 16-pixel wave tiles are all-in or all-out)
#define NATT 300
#define OC   32    // conv1 out channels
#define PIXB 32    // pixels per block in main kernel
#define MST  141   // M row stride in u32 (141 mod 64 = 13 -> conflict-free A reads)
#define EST  18    // entity LDS row stride in floats (16*18 mod 64 = 32 -> halves disjoint)

#if defined(__has_builtin)
#if __has_builtin(__builtin_amdgcn_global_load_async_to_lds_b128) && \
    __has_builtin(__builtin_amdgcn_s_wait_asynccnt)
#define USE_ASYNC_LDS 1
#endif
#endif

typedef __attribute__((ext_vector_type(16))) _Float16 v16h;
typedef __attribute__((ext_vector_type(8)))  float    v8f;
typedef __attribute__((__vector_size__(16))) int      i32x4;

// ---------------- Kernel 1: emb = A1h @ Wemb^T + b ; sx = count_sketch(emb, h1, s1)
__global__ __launch_bounds__(DD)
void k_sketch(const float* __restrict__ a1h, const float* __restrict__ Wemb,
              const float* __restrict__ bemb, const int* __restrict__ h1,
              const float* __restrict__ s1, float* __restrict__ sxg) {
  __shared__ float sx[DD];
  int b = blockIdx.x;
  int c = threadIdx.x;
  sx[c] = 0.f;
  __syncthreads();
  const float* arow = a1h + b * NATT;
  const float* wrow = Wemb + c * NATT;
  float acc = bemb[c];
  #pragma unroll 4
  for (int a = 0; a < NATT; ++a) acc = fmaf(arow[a], wrow[a], acc);
  atomicAdd(&sx[h1[c] & 255], s1[c] * acc);
  __syncthreads();
  sxg[b * DD + c] = sx[c];
}

// ---------------- Kernel 2: G[b,o,m] = sum_j sx[b,j]*w1[o,(j+m)%256];
//                  M[b,o,c] = s2[c]*G[b,o,h2[c]]  (stored f16 for WMMA A-operand)
__global__ __launch_bounds__(DD)
void k_gmat(const float* __restrict__ sxg, const float* __restrict__ w1,
            const int* __restrict__ h2, const float* __restrict__ s2,
            _Float16* __restrict__ Mg) {
  __shared__ float sxl[DD];
  __shared__ float w1l[DD];
  __shared__ float grow[DD];
  int b = blockIdx.x, og = blockIdx.y;
  int t = threadIdx.x;
  sxl[t] = sxg[b * DD + t];
  for (int oo = 0; oo < 4; ++oo) {
    int o = og * 4 + oo;
    __syncthreads();                 // sxl ready / previous grow+w1l readers done
    w1l[t] = w1[o * DD + t];
    __syncthreads();
    float g = 0.f;
    #pragma unroll 4
    for (int j = 0; j < DD; ++j) g = fmaf(sxl[j], w1l[(j + t) & 255], g);
    grow[t] = g;
    __syncthreads();
    Mg[((size_t)b * OC + o) * DD + t] = (_Float16)(s2[t] * grow[h2[t] & 255]);
  }
}

// ---------------- Kernel 3 (fused, WMMA): hidden = relu(M @ entity_tile),
// attr_map = sigmoid(w2 . hidden), attr_feature = attr_map * entity (tile reused from LDS)
__global__ __launch_bounds__(64)
void k_main(const float* __restrict__ ent, const _Float16* __restrict__ Mg,
            const float* __restrict__ w2, float* __restrict__ outMap,
            float* __restrict__ outFeat) {
  __shared__ unsigned Ml[OC * MST];        // A-matrix, f16 pairs, padded rows
  __shared__ float    entw[2][CC * EST];   // per-wave 256x16 f32 tile, padded rows
  __shared__ float    aml[PIXB];

  int b    = blockIdx.y;
  int grp  = blockIdx.x;
  int tid  = threadIdx.x;
  int wv   = tid >> 5;
  int lane = tid & 31;
  int half = lane >> 4;
  int lr   = lane & 15;
  int pbase = grp * PIXB + wv * 16;

  const float* eb = ent + (size_t)b * CC * HW;
  float* ew = entw[wv];

  // ---- Entity tile: 256 rows x 16 f32, b128 per lane; async-to-LDS when available
  if (pbase < HW) {   // whole 16-pixel tile valid (HW is a multiple of 16)
#ifdef USE_ASYNC_LDS
    const __attribute__((address_space(1))) char* g1 =
        (const __attribute__((address_space(1))) char*)(eb + pbase);
    __attribute__((address_space(3))) char* l3 =
        (__attribute__((address_space(3))) char*)ew;
    for (int i = lane; i < CC * 4; i += 32) {
      int c = i >> 2, q = (i & 3) << 2;             // q in {0,4,8,12}
      __builtin_amdgcn_global_load_async_to_lds_b128(
          (__attribute__((address_space(1))) i32x4*)(g1 + ((size_t)c * HW + q) * 4),
          (__attribute__((address_space(3))) i32x4*)(l3 + (c * EST + q) * 4),
          0, 0);
    }
#else
    for (int i = lane; i < CC * 4; i += 32) {
      int c = i >> 2, q = (i & 3) << 2;
      float4 v = *(const float4*)(eb + (size_t)c * HW + pbase + q);
      *(float2*)&ew[c * EST + q]     = make_float2(v.x, v.y);
      *(float2*)&ew[c * EST + q + 2] = make_float2(v.z, v.w);
    }
#endif
  } else {            // fully out-of-bounds wave tile: zero-fill
    for (int i = lane; i < CC * 4; i += 32) {
      int c = i >> 2, q = (i & 3) << 2;
      *(float2*)&ew[c * EST + q]     = make_float2(0.f, 0.f);
      *(float2*)&ew[c * EST + q + 2] = make_float2(0.f, 0.f);
    }
  }

  // ---- M tile: 32x256 f16 = 1024 uint4, coalesced b128 loads (overlaps async DMA)
  const uint4* Mu4 = (const uint4*)(Mg + (size_t)b * OC * DD);
  for (int i = tid; i < OC * 32; i += 64) {
    uint4 m4 = Mu4[i];
    unsigned* dst = &Ml[(i >> 5) * MST + ((i & 31) << 2)];
    dst[0] = m4.x; dst[1] = m4.y; dst[2] = m4.z; dst[3] = m4.w;
  }

#ifdef USE_ASYNC_LDS
  __builtin_amdgcn_s_wait_asynccnt(0);
#endif
  __syncthreads();

  v8f acc0 = {}; v8f acc1 = {};
  #pragma unroll
  for (int ks = 0; ks < 8; ++ks) {
    int kb = ks * 32;
    // B operand: B[k][n], lane half selects K group, f32->f16 on the fly
    union { unsigned u[8]; v16h v; } bop;
    #pragma unroll
    for (int v = 0; v < 8; ++v) {
      int k0 = kb + half * 16 + 2 * v;
      _Float16 h0 = (_Float16)ew[k0 * EST + lr];
      _Float16 h1 = (_Float16)ew[(k0 + 1) * EST + lr];
      bop.u[v] = (unsigned)__builtin_bit_cast(unsigned short, h0) |
                 ((unsigned)__builtin_bit_cast(unsigned short, h1) << 16);
    }
    // A operands: rows o=lr (+16 for second tile), lane half selects K group (same as B)
    union { unsigned u[8]; v16h v; } a0, a1;
    #pragma unroll
    for (int v = 0; v < 8; ++v) {
      int k0 = 2 * v + 16 * half;
      int cu = (kb + k0) >> 1;
      a0.u[v] = Ml[lr * MST + cu];
      a1.u[v] = Ml[(16 + lr) * MST + cu];
    }
    acc0 = __builtin_amdgcn_wmma_f32_16x16x32_f16(false, a0.v, false, bop.v,
                                                  (short)0, acc0, false, false);
    acc1 = __builtin_amdgcn_wmma_f32_16x16x32_f16(false, a1.v, false, bop.v,
                                                  (short)0, acc1, false, false);
  }

  // relu + conv2 (32->1) + cross-half reduce + sigmoid
  float s = 0.f;
  #pragma unroll
  for (int r = 0; r < 8; ++r) {
    s = fmaf(fmaxf(acc0[r], 0.f), w2[8 * half + r], s);
    s = fmaf(fmaxf(acc1[r], 0.f), w2[16 + 8 * half + r], s);
  }
  s += __shfl_xor(s, 16, 32);
  float am = 1.f / (1.f + __expf(-s));
  int p = pbase + lr;
  if (half == 0) {
    aml[wv * 16 + lr] = am;
    if (p < HW) outMap[b * HW + p] = am;
  }
  __syncthreads();

  // attr_feature = attr_map * entity: LDS tile reused, b128 stores (512 B / wave instr)
  float* ob = outFeat + (size_t)b * CC * HW + grp * PIXB;
  for (int i = tid; i < (CC * PIXB) / 4; i += 64) {
    int c = i >> 3, q = (i & 7) << 2;               // q in {0,4,...,28}
    if (grp * PIXB + q + 3 < HW) {
      const float* et = entw[q >> 4] + c * EST + (q & 15);
      float4 v = make_float4(aml[q] * et[0], aml[q + 1] * et[1],
                             aml[q + 2] * et[2], aml[q + 3] * et[3]);
      *(float4*)&ob[(size_t)c * HW + q] = v;
    }
  }
}

extern "C" void kernel_launch(void* const* d_in, const int* in_sizes, int n_in,
                              void* d_out, int out_size, void* d_ws, size_t ws_size,
                              hipStream_t stream) {
  (void)in_sizes; (void)n_in; (void)out_size; (void)ws_size;
  const float* ent  = (const float*)d_in[0];
  const float* a1h  = (const float*)d_in[1];
  const float* Wemb = (const float*)d_in[2];
  const float* bemb = (const float*)d_in[3];
  const int*   h1   = (const int*)d_in[4];
  const float* s1   = (const float*)d_in[5];
  const int*   h2   = (const int*)d_in[6];
  const float* s2   = (const float*)d_in[7];
  const float* w1   = (const float*)d_in[8];
  const float* w2   = (const float*)d_in[9];

  float*    sxg = (float*)d_ws;                                       // 64*256 f32
  _Float16* Mg  = (_Float16*)((char*)d_ws + BB * DD * sizeof(float)); // 64*32*256 f16 (16B-aligned)

  float* outMap  = (float*)d_out;        // [64,1,28,28]
  float* outFeat = outMap + BB * HW;     // [64,256,28,28]

  k_sketch<<<dim3(BB), dim3(DD), 0, stream>>>(a1h, Wemb, bemb, h1, s1, sxg);
  k_gmat  <<<dim3(BB, 8), dim3(DD), 0, stream>>>(sxg, w1, h2, s2, Mg);
  k_main  <<<dim3((HW + PIXB - 1) / PIXB, BB), dim3(64), 0, stream>>>(ent, Mg, w2, outMap, outFeat);
}
